// MHA_91173565759554
// MI455X (gfx1250) — compile-verified, matrix operand order checked by reading, not run on hardware
//
#include <hip/hip_runtime.h>

typedef __attribute__((ext_vector_type(16))) __bf16 v16bf;
typedef __attribute__((ext_vector_type(8)))  float  v8f;

#define D_MODEL 2048
#define N_HEADS 32
#define D_HEAD  64
#define BATCH   2
#define SEQ     2048
#define MTOT    (BATCH * SEQ)        // 4096 rows
#define NQKV    (N_HEADS * D_HEAD)   // 2048

static __device__ __forceinline__ v16bf ld16(const __bf16* p) {
  return *(const v16bf*)p;
}

// ---- CDNA5 async global->LDS copy (16 bytes per lane), tracked by ASYNCcnt ----
static __device__ __forceinline__ void async_b128(void* lds, const void* gptr) {
  unsigned           l = (unsigned)(unsigned long long)lds;   // LDS byte offset
  unsigned long long g = (unsigned long long)gptr;
  asm volatile("global_load_async_to_lds_b128 %0, %1, off"
               :: "v"(l), "v"(g) : "memory");
}
static __device__ __forceinline__ void wait_async_le4() {
  asm volatile("s_wait_asynccnt 0x4" ::: "memory");
}
static __device__ __forceinline__ void wait_async_0() {
  asm volatile("s_wait_asynccnt 0x0" ::: "memory");
}

// ---------------- elementwise f32 -> bf16 ----------------
__global__ void cvt_bf16_kernel(const float* __restrict__ in,
                                __bf16* __restrict__ out, int n) {
  int i = blockIdx.x * blockDim.x + threadIdx.x;
  if (i < n) out[i] = (__bf16)in[i];
}

// ------------- transpose + convert: Wt[n][k] = W[k][n] -------------
__global__ void transpose_cvt_kernel(const float* __restrict__ W,
                                     __bf16* __restrict__ Wt, int K, int N) {
  int i = blockIdx.x * blockDim.x + threadIdx.x;
  if (i >= K * N) return;
  int n = i / K, k = i - n * K;
  Wt[i] = (__bf16)W[(size_t)k * N + n];
}

// ---------------- bf16 WMMA GEMM: C(MxN) = A(MxK) * Bt(NxK)^T ----------------
// 128x128 block, BK=32, 8 waves (each 32x64 patch = 8 wmma/step),
// double-buffered LDS fed by async global->LDS loads.
#define BM 128
#define BN 128
#define BK 32

__launch_bounds__(256)
__global__ void gemm_bf16_kernel(const __bf16* __restrict__ A,
                                 const __bf16* __restrict__ Bt,
                                 float* __restrict__ C,
                                 int M, int N, int K) {
  __shared__ __align__(32) __bf16 As[2][BM][BK];
  __shared__ __align__(32) __bf16 Bs[2][BN][BK];
  const int tid   = threadIdx.x;
  const int lane  = tid & 31;
  const int wave  = tid >> 5;
  const int m0    = blockIdx.y * BM;
  const int n0    = blockIdx.x * BN;
  const int wm    = (wave & 3) * 32;   // 4 waves down M
  const int wn    = (wave >> 2) * 64;  // 2 waves across N
  const int lr    = lane & 15;
  const int kh    = (lane >> 4) * 16;
  const int ldrow = tid >> 1;          // 0..127
  const int ldcol = (tid & 1) * 16;    // 0 or 16 (bf16 units)

  const __bf16* agp = &A [(size_t)(m0 + ldrow) * K + ldcol];
  const __bf16* bgp = &Bt[(size_t)(n0 + ldrow) * K + ldcol];

  v8f acc[2][4] = {};

  // fire tile 0 into buffer 0 (4 async b128 per wave per tile)
  async_b128(&As[0][ldrow][ldcol],     agp);
  async_b128(&As[0][ldrow][ldcol] + 8, agp + 8);
  async_b128(&Bs[0][ldrow][ldcol],     bgp);
  async_b128(&Bs[0][ldrow][ldcol] + 8, bgp + 8);

  int buf = 0;
  for (int k0 = 0; k0 < K; k0 += BK) {
    if (k0 + BK < K) {
      // prefetch next tile into the other buffer while we compute
      const int nb = buf ^ 1;
      async_b128(&As[nb][ldrow][ldcol],     agp + k0 + BK);
      async_b128(&As[nb][ldrow][ldcol] + 8, agp + k0 + BK + 8);
      async_b128(&Bs[nb][ldrow][ldcol],     bgp + k0 + BK);
      async_b128(&Bs[nb][ldrow][ldcol] + 8, bgp + k0 + BK + 8);
      wait_async_le4();   // current tile's 4 ops complete (in-order), prefetch in flight
    } else {
      wait_async_0();
    }
    __syncthreads();

    v16bf a0 = ld16(&As[buf][wm + lr][kh]);
    v16bf a1 = ld16(&As[buf][wm + 16 + lr][kh]);
    v16bf b[4];
#pragma unroll
    for (int ni = 0; ni < 4; ++ni) b[ni] = ld16(&Bs[buf][wn + ni * 16 + lr][kh]);
#pragma unroll
    for (int ni = 0; ni < 4; ++ni) {
      acc[0][ni] = __builtin_amdgcn_wmma_f32_16x16x32_bf16(
          false, a0, false, b[ni], (short)0, acc[0][ni], false, false);
      acc[1][ni] = __builtin_amdgcn_wmma_f32_16x16x32_bf16(
          false, a1, false, b[ni], (short)0, acc[1][ni], false, false);
    }
    __syncthreads();   // all reads of buf done before it is overwritten next round
    buf ^= 1;
  }

  const int rbase = (lane >> 4) * 8;
#pragma unroll
  for (int mi = 0; mi < 2; ++mi)
#pragma unroll
    for (int ni = 0; ni < 4; ++ni)
#pragma unroll
      for (int r = 0; r < 8; ++r) {
        int row = m0 + wm + mi * 16 + rbase + r;
        int col = n0 + wn + ni * 16 + lr;
        C[(size_t)row * N + col] = acc[mi][ni][r];
      }
}

// ---------------- RoPE: rotate Q (ws) and K (in-place in d_out), emit bf16 [b][h][t][d] ----------------
__global__ void rope_kernel(const float* __restrict__ qin,
                            float* __restrict__ kio,
                            const float* __restrict__ freqs,
                            __bf16* __restrict__ qbf,
                            __bf16* __restrict__ kbf) {
  const int HALF = D_HEAD / 2; // 32 freq pairs
  int i = blockIdx.x * blockDim.x + threadIdx.x;
  int total = BATCH * SEQ * N_HEADS * HALF;
  if (i >= total) return;
  int p = i % HALF;
  int h = (i / HALF) % N_HEADS;
  int t = (i / (HALF * N_HEADS)) % SEQ;
  int b = i / (HALF * N_HEADS * SEQ);
  float f = freqs[t * HALF + p];
  float c = cosf(f), s = sinf(f);
  size_t src = (size_t)(b * SEQ + t) * NQKV + h * D_HEAD + 2 * p;
  size_t dst = ((size_t)(b * N_HEADS + h) * SEQ + t) * D_HEAD + 2 * p;
  float q0 = qin[src], q1 = qin[src + 1];
  qbf[dst]     = (__bf16)(q0 * c - q1 * s);
  qbf[dst + 1] = (__bf16)(q0 * s + q1 * c);
  float k0 = kio[src], k1 = kio[src + 1];
  float ko0 = k0 * c - k1 * s;
  float ko1 = k0 * s + k1 * c;
  kio[src]     = ko0;           // k output is post-RoPE in the reference
  kio[src + 1] = ko1;
  kbf[dst]     = (__bf16)ko0;
  kbf[dst + 1] = (__bf16)ko1;
}

// ---------------- V transpose to bf16 [b][h][d][t] ----------------
__global__ void vtrans_kernel(const float* __restrict__ v, __bf16* __restrict__ vt) {
  int i = blockIdx.x * blockDim.x + threadIdx.x;
  if (i >= MTOT * NQKV) return;
  int n = i % NQKV;
  int t = (i / NQKV) % SEQ;
  int b = i / (NQKV * SEQ);
  int h = n / D_HEAD, d = n % D_HEAD;
  vt[((size_t)(b * N_HEADS + h) * D_HEAD + d) * SEQ + t] = (__bf16)v[i];
}

// ---------------- streaming-softmax attention, all matmuls via WMMA ----------------
#define AW 4  // waves per block, one 16-query tile each

__launch_bounds__(128)
__global__ void attn_kernel(const __bf16* __restrict__ Qbf,
                            const __bf16* __restrict__ Kbf,
                            const __bf16* __restrict__ Vt,
                            __bf16* __restrict__ Obf) {
  __shared__ __align__(32) float  Stile[AW][16][32];
  __shared__ __align__(32) __bf16 Pbf[AW][16][32];
  __shared__ float rowM[AW][16], rowL[AW][16], rowF[AW][16];

  const int lane  = threadIdx.x & 31;
  const int wave  = threadIdx.x >> 5;
  const int bh    = blockIdx.y;                       // b*N_HEADS + h
  const int m0    = (blockIdx.x * AW + wave) * 16;    // query tile
  const int lr    = lane & 15;
  const int kh    = (lane >> 4) * 16;
  const int rbase = (lane >> 4) * 8;

  const __bf16* Qb = Qbf + (size_t)bh * SEQ * D_HEAD;
  const __bf16* Kb = Kbf + (size_t)bh * SEQ * D_HEAD;
  const __bf16* Vb = Vt  + (size_t)bh * D_HEAD * SEQ;

  // Q fragments (16x64 split as two 16x32 A fragments), held for the whole pass
  v16bf aq0 = ld16(&Qb[(size_t)(m0 + lr) * D_HEAD + kh]);
  v16bf aq1 = ld16(&Qb[(size_t)(m0 + lr) * D_HEAD + 32 + kh]);

  v8f acc[4] = {};   // O accumulator: 16x64 f32 (4 C tiles)
  if (lane < 16) { rowM[wave][lane] = -3.0e38f; rowL[wave][lane] = 0.f; }
  __syncthreads();

  for (int s0 = 0; s0 < SEQ; s0 += 32) {
    // S = Q Kᵀ for a 16x32 key block (K rows are the B-matrix columns)
#pragma unroll
    for (int j = 0; j < 2; ++j) {
      v16bf bk0 = ld16(&Kb[(size_t)(s0 + j * 16 + lr) * D_HEAD + kh]);
      v16bf bk1 = ld16(&Kb[(size_t)(s0 + j * 16 + lr) * D_HEAD + 32 + kh]);
      v8f st = {};
      st = __builtin_amdgcn_wmma_f32_16x16x32_bf16(false, aq0, false, bk0,
                                                   (short)0, st, false, false);
      st = __builtin_amdgcn_wmma_f32_16x16x32_bf16(false, aq1, false, bk1,
                                                   (short)0, st, false, false);
#pragma unroll
      for (int r = 0; r < 8; ++r)
        Stile[wave][rbase + r][j * 16 + lr] = st[r] * 0.125f;  // 1/sqrt(64)
    }
    __syncthreads();

    // streaming softmax stats per row (lanes 0..15, one row each)
    if (lane < 16) {
      float mold = rowM[wave][lane];
      float mx = mold;
#pragma unroll
      for (int c = 0; c < 32; ++c) mx = fmaxf(mx, Stile[wave][lane][c]);
      float fac = __expf(mold - mx);
      float sum = 0.f;
#pragma unroll
      for (int c = 0; c < 32; ++c) {
        float pv = __expf(Stile[wave][lane][c] - mx);
        sum += pv;
        Pbf[wave][lane][c] = (__bf16)pv;
      }
      rowL[wave][lane] = rowL[wave][lane] * fac + sum;
      rowM[wave][lane] = mx;
      rowF[wave][lane] = fac;
    }
    __syncthreads();

    // rescale running O, then O += P @ V  (P is a 16x32 bf16 A fragment)
    float f[8];
#pragma unroll
    for (int r = 0; r < 8; ++r) f[r] = rowF[wave][rbase + r];
    v16bf ap = ld16(&Pbf[wave][lr][kh]);
#pragma unroll
    for (int ni = 0; ni < 4; ++ni) {
#pragma unroll
      for (int r = 0; r < 8; ++r) acc[ni][r] *= f[r];
      v16bf bv = ld16(&Vb[(size_t)(ni * 16 + lr) * SEQ + s0 + kh]);
      acc[ni] = __builtin_amdgcn_wmma_f32_16x16x32_bf16(
          false, ap, false, bv, (short)0, acc[ni], false, false);
    }
    __syncthreads();
  }

  // normalize and emit bf16 [b][t][h*64+d] (A layout for the Wo GEMM)
  float invl[8];
#pragma unroll
  for (int r = 0; r < 8; ++r) invl[r] = 1.0f / rowL[wave][rbase + r];
  const int b = bh >> 5;
  const int h = bh & 31;
#pragma unroll
  for (int ni = 0; ni < 4; ++ni)
#pragma unroll
    for (int r = 0; r < 8; ++r) {
      int row = m0 + rbase + r;
      int col = h * D_HEAD + ni * 16 + lr;
      Obf[(size_t)(b * SEQ + row) * NQKV + col] = (__bf16)(acc[ni][r] * invl[r]);
    }
}

// ---------------- host-side orchestration ----------------
extern "C" void kernel_launch(void* const* d_in, const int* in_sizes, int n_in,
                              void* d_out, int out_size, void* d_ws, size_t ws_size,
                              hipStream_t stream) {
  (void)in_sizes; (void)n_in; (void)out_size; (void)ws_size;
  const float* x     = (const float*)d_in[0];
  const float* Wq    = (const float*)d_in[1];
  const float* Wk    = (const float*)d_in[2];
  const float* Wv    = (const float*)d_in[3];
  const float* Wo    = (const float*)d_in[4];
  const float* freqs = (const float*)d_in[5];

  float* out   = (float*)d_out;                       // [B,T,D_MODEL]
  float* out_k = out   + (size_t)MTOT * D_MODEL;      // [B,T,H,D]
  float* out_v = out_k + (size_t)MTOT * NQKV;         // [B,T,H,D]

  char* ws = (char*)d_ws;
  const size_t MB = 1ull << 20;
  __bf16* xbf = (__bf16*)(ws);             // 16 MB  x bf16 [4096][2048]
  __bf16* WqT = (__bf16*)(ws + 16 * MB);   //  8 MB  Wq^T bf16
  __bf16* WkT = (__bf16*)(ws + 24 * MB);
  __bf16* WvT = (__bf16*)(ws + 32 * MB);
  __bf16* WoT = (__bf16*)(ws + 40 * MB);
  float*  Qf  = (float*) (ws + 48 * MB);   // 32 MB  Q f32 pre-RoPE
  __bf16* Qbf = (__bf16*)(ws + 80 * MB);   // 16 MB  Q bf16 [b][h][t][d]
  __bf16* Kbf = (__bf16*)(ws + 96 * MB);   // 16 MB  K bf16 [b][h][t][d]
  __bf16* Vtb = (__bf16*)(ws + 112 * MB);  // 16 MB  V bf16 [b][h][d][t]
  __bf16* Obf = (__bf16*)(ws + 128 * MB);  // 16 MB  attn out bf16

  const int nx = MTOT * D_MODEL;
  cvt_bf16_kernel<<<(nx + 255) / 256, 256, 0, stream>>>(x, xbf, nx);
  const int nw = D_MODEL * NQKV;
  transpose_cvt_kernel<<<(nw + 255) / 256, 256, 0, stream>>>(Wq, WqT, D_MODEL, NQKV);
  transpose_cvt_kernel<<<(nw + 255) / 256, 256, 0, stream>>>(Wk, WkT, D_MODEL, NQKV);
  transpose_cvt_kernel<<<(nw + 255) / 256, 256, 0, stream>>>(Wv, WvT, D_MODEL, NQKV);
  transpose_cvt_kernel<<<(nw + 255) / 256, 256, 0, stream>>>(Wo, WoT, NQKV, D_MODEL);

  dim3 gg(NQKV / BN, MTOT / BM);
  gemm_bf16_kernel<<<gg, 256, 0, stream>>>(xbf, WqT, Qf,    MTOT, NQKV, D_MODEL);
  gemm_bf16_kernel<<<gg, 256, 0, stream>>>(xbf, WkT, out_k, MTOT, NQKV, D_MODEL);
  gemm_bf16_kernel<<<gg, 256, 0, stream>>>(xbf, WvT, out_v, MTOT, NQKV, D_MODEL);

  const int nr = BATCH * SEQ * N_HEADS * (D_HEAD / 2);
  rope_kernel<<<(nr + 255) / 256, 256, 0, stream>>>(Qf, out_k, freqs, Qbf, Kbf);
  vtrans_kernel<<<(nx + 255) / 256, 256, 0, stream>>>(out_v, Vtb);

  dim3 ga(SEQ / (16 * AW), BATCH * N_HEADS);
  attn_kernel<<<ga, 32 * AW, 0, stream>>>(Qbf, Kbf, Vtb, Obf);

  dim3 go(D_MODEL / BN, MTOT / BM);
  gemm_bf16_kernel<<<go, 256, 0, stream>>>(Obf, WoT, out, MTOT, D_MODEL, NQKV);
}
